// VectorDecoder_90013924589786
// MI455X (gfx1250) — matrix operand order repaired and checked by reading, not compiled, not run on hardware
//
#include <hip/hip_runtime.h>

typedef __attribute__((ext_vector_type(16))) _Float16 v16h;
typedef __attribute__((ext_vector_type(8)))  _Float16 v8h;
typedef __attribute__((ext_vector_type(8)))  float    v8f;

// ---------------- constants ----------------
constexpr int CB   = 16;      // batch
constexpr int NL   = 55;      // lanes
constexpr int NM   = 256;     // mid
constexpr int NC   = 12000;   // coordinates
constexpr int CH   = 128;     // channels
// d_in indices (setup_inputs insertion order, params flattened in insertion order)
// 0 hlane 1 hmid 2 hinteraction 3 coordinates 4 c_mask 5 masker
// ls_attn: 6 Wq 7 bq 8 Wk 9 bk 10 Wv 11 bv 12 Wo 13 bo
// connect: 14 W1 15 b1 16 g 17 be 18 Wf 19 bf
// ego2coor: 20 W 21 b 22 g 23 be
// l2c:  24..31 (Wq bq Wk bk Wv bv Wo bo)
// l2c2: 32..39
// convert: 40 W1 41 b1 42 g 43 be 44 Wf 45 bf

// ---------------- helpers ----------------
__device__ __forceinline__ int kmapA(int l, int e) { return ((l >> 4) << 3) + (e < 8 ? e : e + 8); }
__device__ __forceinline__ int kmapB(int l, int e) { return ((l >> 4) << 3) + (e < 8 ? e : e + 8); }

__device__ __forceinline__ v8f vzero8() {
  v8f z;
#pragma unroll
  for (int i = 0; i < 8; ++i) z[i] = 0.f;
  return z;
}

__device__ __forceinline__ v8f wmma16(v16h a, v16h b, v8f c) {
  return __builtin_amdgcn_wmma_f32_16x16x32_f16(false, a, false, b, (short)0, c, false, false);
}

// A fragment (16x32 f16) from row-major [rows x ld] at column offset folded into base.
__device__ __forceinline__ v16h load_a_frag(const _Float16* tile, int ld) {
  int l = threadIdx.x & 31;
  const _Float16* p = tile + (long)(l & 15) * ld + ((l >> 4) << 3);
  v8h lo = *(const v8h*)p;
  v8h hi = *(const v8h*)(p + 16);
  v16h r;
#pragma unroll
  for (int e = 0; e < 8; ++e) { r[e] = lo[e]; r[e + 8] = hi[e]; }
  return r;
}

__device__ __forceinline__ v16h load_a_clamped(const _Float16* base, int ld, int row0,
                                               int Mreal, int k0) {
  int l = threadIdx.x & 31;
  int r = row0 + (l & 15);
  if (r > Mreal - 1) r = Mreal - 1;
  const _Float16* p = base + (long)r * ld + k0 + ((l >> 4) << 3);
  v8h lo = *(const v8h*)p;
  v8h hi = *(const v8h*)(p + 16);
  v16h out;
#pragma unroll
  for (int e = 0; e < 8; ++e) { out[e] = lo[e]; out[e + 8] = hi[e]; }
  return out;
}

// ---------------- weight packing: f32 (K x N) row-major -> B-frag layout ----------------
// frag(kt,nt) base = ((kt*(N/16))+nt)*512 halfs ; element = W[kt*32+kmapB(l,e)][nt*16+l%16]
__global__ void pack_b_kernel(const float* __restrict__ W, _Float16* __restrict__ out,
                              int K, int N) {
  int NT = N >> 4;
  int tile = blockIdx.x;
  int kt = tile / NT, nt = tile % NT;
  int l = threadIdx.x;
  int col = nt * 16 + (l & 15);
  _Float16* dst = out + (long)tile * 512 + l * 16;
#pragma unroll
  for (int e = 0; e < 16; ++e) {
    int k = kt * 32 + kmapB(l, e);
    dst[e] = (_Float16)W[(long)k * N + col];
  }
}

// ---------------- per-batch prep ----------------
__global__ void prep_batch_kernel(const float* __restrict__ hinter,
                                  const float* __restrict__ e2cW, const float* __restrict__ e2cB,
                                  const float* __restrict__ cvW1, const float* __restrict__ cvB1,
                                  const float* __restrict__ cvWf, const float* __restrict__ cvBf,
                                  const float* __restrict__ cnW1, const float* __restrict__ cnB1,
                                  const float* __restrict__ cnWf, const float* __restrict__ cnBf,
                                  float* __restrict__ ego,
                                  float* __restrict__ cvecE2c, float* __restrict__ cvecConv,
                                  float* __restrict__ cvecConn,
                                  float* __restrict__ scalConv, float* __restrict__ scalConn) {
  int b = blockIdx.x, c = threadIdx.x;  // 128 threads
  __shared__ float eg[CH];
  __shared__ float red[CH];
  float e = hinter[((long)b * 56 + 55) * CH + c];
  eg[c] = e;
  ego[b * CH + c] = e;
  __syncthreads();
  float a1 = e2cB[c], a2 = cvB1[c], a3 = cnB1[c];
  for (int k = 0; k < CH; ++k) {
    float ek = eg[k];
    a1 += ek * e2cW[(2 + k) * CH + c];
    a2 += ek * cvW1[k * CH + c];
    a3 += ek * cnW1[k * CH + c];
  }
  cvecE2c[b * CH + c] = a1;
  cvecConv[b * CH + c] = a2;
  cvecConn[b * CH + c] = a3;
  red[c] = e * cvWf[c];
  __syncthreads();
  for (int s = 64; s > 0; s >>= 1) { if (c < s) red[c] += red[c + s]; __syncthreads(); }
  if (c == 0) scalConv[b] = red[0] + cvBf[0];
  __syncthreads();
  red[c] = e * cnWf[c];
  __syncthreads();
  for (int s = 64; s > 0; s >>= 1) { if (c < s) red[c] += red[c + s]; __syncthreads(); }
  if (c == 0) scalConn[b] = red[0] + cnBf[0];
}

__global__ void f32_to_f16_kernel(const float* __restrict__ src, _Float16* __restrict__ dst,
                                  long n) {
  long i = (long)blockIdx.x * blockDim.x + threadIdx.x;
  if (i < n) dst[i] = (_Float16)src[i];
}

// ---------------- pos1 = relu(LN(x*W0 + y*W1 + cvec)) ----------------
__global__ void pos1_kernel(const float* __restrict__ coords, const float* __restrict__ cvec,
                            const float* __restrict__ W,  // ego2coor.W rows 0..1
                            const float* __restrict__ g, const float* __restrict__ be,
                            _Float16* __restrict__ pos1) {
  int wv = threadIdx.x >> 5, l = threadIdx.x & 31;
  long r = (long)blockIdx.x * (blockDim.x >> 5) + wv;
  if (r >= (long)CB * NC) return;
  int b = (int)(r / NC), i = (int)(r % NC);
  float x = coords[i * 2], y = coords[i * 2 + 1];
  float v[4];
#pragma unroll
  for (int j = 0; j < 4; ++j) {
    int c = l + j * 32;
    v[j] = x * W[c] + y * W[CH + c] + cvec[b * CH + c];
  }
  float s = v[0] + v[1] + v[2] + v[3];
  for (int m = 16; m > 0; m >>= 1) s += __shfl_xor(s, m, 32);
  float mu = s * (1.f / 128.f);
  float q = 0.f;
#pragma unroll
  for (int j = 0; j < 4; ++j) { float d = v[j] - mu; q += d * d; }
  for (int m = 16; m > 0; m >>= 1) q += __shfl_xor(q, m, 32);
  float rstd = rsqrtf(q * (1.f / 128.f) + 1e-5f);
#pragma unroll
  for (int j = 0; j < 4; ++j) {
    int c = l + j * 32;
    float h = (v[j] - mu) * rstd * g[c] + be[c];
    h = h > 0.f ? h : 0.f;
    pos1[((long)b * NC + i) * CH + c] = (_Float16)h;
  }
}

// ---------------- KV projection + pack into B-frag layouts ----------------
// Kpack per batch: [h][kt2(2)][nt(NKT)] frags of K^T (64 x NKA):  val = K[nt*16+l%16][h*64+kt2*32+kmapB]
// Vpack per batch: [h][ktv(NKA/32)][ntv(4)] frags of V (NKA x 64): val = V[ktv*32+kmapB][h*64+ntv*16+l%16]
template <int NKT>
__global__ void __launch_bounds__(128) kvproj_kernel(
    const _Float16* __restrict__ X, long xStride, int Mreal,
    const _Float16* __restrict__ WkP, const float* __restrict__ bk,
    const _Float16* __restrict__ WvP, const float* __restrict__ bv,
    _Float16* __restrict__ Kpack, _Float16* __restrict__ Vpack) {
  constexpr int NKA = NKT * 16;
  constexpr int KTV = NKA / 32;
  int b = blockIdx.y;
  int w = threadIdx.x >> 5, l = threadIdx.x & 31;
  int col = l & 15, half = l >> 4;
  int keyBase = blockIdx.x * 64;
  __shared__ _Float16 Kl[64 * CH];
  __shared__ _Float16 Vl[64 * CH];
  const _Float16* Xb = X + (long)b * xStride;
  int row0 = keyBase + w * 16;

  v8f accK[8], accV[8];
#pragma unroll
  for (int nt = 0; nt < 8; ++nt) { accK[nt] = vzero8(); accV[nt] = vzero8(); }
#pragma unroll
  for (int kt = 0; kt < 4; ++kt) {
    v16h a = load_a_clamped(Xb, CH, row0, Mreal, kt * 32);
#pragma unroll
    for (int nt = 0; nt < 8; ++nt) {
      v16h bk_ = *(const v16h*)(WkP + (long)(kt * 8 + nt) * 512 + l * 16);
      accK[nt] = wmma16(a, bk_, accK[nt]);
      v16h bv_ = *(const v16h*)(WvP + (long)(kt * 8 + nt) * 512 + l * 16);
      accV[nt] = wmma16(a, bv_, accV[nt]);
    }
  }
#pragma unroll
  for (int nt = 0; nt < 8; ++nt) {
    float bkc = bk[nt * 16 + col], bvc = bv[nt * 16 + col];
#pragma unroll
    for (int r = 0; r < 8; ++r) {
      int rr = w * 16 + r + half * 8;
      Kl[rr * CH + nt * 16 + col] = (_Float16)(accK[nt][r] + bkc);
      Vl[rr * CH + nt * 16 + col] = (_Float16)(accV[nt][r] + bvc);
    }
  }
  __syncthreads();
  // pack K: one key-tile (nt) per wave
  {
    int nt = (keyBase >> 4) + w;
    if (nt < NKT) {
      for (int h = 0; h < 2; ++h)
        for (int kt2 = 0; kt2 < 2; ++kt2) {
          _Float16* dst = Kpack + ((long)b * (2 * 2 * NKT) + (long)(h * 2 + kt2) * NKT + nt) * 512 + l * 16;
          int key = w * 16 + (l & 15);
#pragma unroll
          for (int e = 0; e < 16; ++e) {
            int d = h * 64 + kt2 * 32 + kmapB(l, e);
            dst[e] = Kl[key * CH + d];
          }
        }
    }
  }
  // pack V: 16 frags/block, 4 per wave
  for (int f = w; f < 16; f += 4) {
    int h = f & 1, ktvl = (f >> 1) & 1, ntv = f >> 2;
    int ktv = (keyBase >> 5) + ktvl;
    if (ktv < KTV) {
      _Float16* dst = Vpack + ((long)b * (2 * KTV * 4) + (long)(h * KTV + ktv) * 4 + ntv) * 512 + l * 16;
#pragma unroll
      for (int e = 0; e < 16; ++e) {
        int key = ktvl * 32 + kmapB(l, e);
        int n = h * 64 + ntv * 16 + (l & 15);
        dst[e] = Vl[key * CH + n];
      }
    }
  }
}

// ---------------- fused attention: Q=X@Wq+bq ; S=Q K^T/8 masked ; P=softmax ; O=PV ; out=O@Wo+bo
template <int NKT>
__global__ void __launch_bounds__(128) attn_kernel(
    const _Float16* __restrict__ Xq, long xStride, int M,
    const _Float16* __restrict__ WqP, const float* __restrict__ bq,
    const _Float16* __restrict__ Kp, const _Float16* __restrict__ Vp,
    const float* __restrict__ mask, int maskLd, int NkReal,
    const _Float16* __restrict__ WoP, const float* __restrict__ bo,
    _Float16* __restrict__ Out, long oStride) {
  constexpr int NKA = NKT * 16;
  constexpr int KTV = NKA / 32;
  int b = blockIdx.y;
  int w = threadIdx.x >> 5, l = threadIdx.x & 31;
  int col = l & 15, half = l >> 4;
  int row0 = blockIdx.x * 64 + w * 16;
  __shared__ float maskS[NKA];
  __shared__ _Float16 Qs[4][16 * CH];
  __shared__ _Float16 Ps[4][16 * NKA];
  for (int n = threadIdx.x; n < NKA; n += blockDim.x)
    maskS[n] = (n < NkReal) ? mask[(long)b * maskLd + n] : 0.f;
  __syncthreads();
  if (row0 >= M) return;  // no block barriers after this point
  const _Float16* Xb = Xq + (long)b * xStride;

  {  // Q projection
    v8f acc[8];
#pragma unroll
    for (int nt = 0; nt < 8; ++nt) acc[nt] = vzero8();
#pragma unroll
    for (int kt = 0; kt < 4; ++kt) {
      v16h a = load_a_clamped(Xb, CH, row0, M, kt * 32);
#pragma unroll
      for (int nt = 0; nt < 8; ++nt) {
        v16h bf = *(const v16h*)(WqP + (long)(kt * 8 + nt) * 512 + l * 16);
        acc[nt] = wmma16(a, bf, acc[nt]);
      }
    }
#pragma unroll
    for (int nt = 0; nt < 8; ++nt) {
      float bb = bq[nt * 16 + col];
#pragma unroll
      for (int r = 0; r < 8; ++r)
        Qs[w][(r + half * 8) * CH + nt * 16 + col] = (_Float16)(acc[nt][r] + bb);
    }
  }

  v8f o[8];
#pragma unroll
  for (int i = 0; i < 8; ++i) o[i] = vzero8();

  for (int h = 0; h < 2; ++h) {
    v8f s[NKT];
#pragma unroll
    for (int nt = 0; nt < NKT; ++nt) s[nt] = vzero8();
#pragma unroll
    for (int kt2 = 0; kt2 < 2; ++kt2) {
      v16h a = load_a_frag(&Qs[w][0] + h * 64 + kt2 * 32, CH);
#pragma unroll
      for (int nt = 0; nt < NKT; ++nt) {
        v16h bf = *(const v16h*)(Kp + ((long)b * (2 * 2 * NKT) + (long)(h * 2 + kt2) * NKT + nt) * 512 + l * 16);
        s[nt] = wmma16(a, bf, s[nt]);
      }
    }
    float mx[8], sm[8];
#pragma unroll
    for (int r = 0; r < 8; ++r) { mx[r] = -3.0e38f; sm[r] = 0.f; }
#pragma unroll
    for (int nt = 0; nt < NKT; ++nt) {
      int n = nt * 16 + col;
      float pen = (n < NkReal) ? (-1.0e7f * (1.f - maskS[n])) : -3.0e38f;
#pragma unroll
      for (int r = 0; r < 8; ++r) {
        float sv = s[nt][r] * 0.125f + pen;  // 1/sqrt(64)
        s[nt][r] = sv;
        mx[r] = fmaxf(mx[r], sv);
      }
    }
#pragma unroll
    for (int r = 0; r < 8; ++r)
      for (int m = 1; m < 16; m <<= 1) mx[r] = fmaxf(mx[r], __shfl_xor(mx[r], m, 32));
#pragma unroll
    for (int nt = 0; nt < NKT; ++nt)
#pragma unroll
      for (int r = 0; r < 8; ++r) {
        float p = __expf(s[nt][r] - mx[r]);
        s[nt][r] = p;
        sm[r] += p;
      }
#pragma unroll
    for (int r = 0; r < 8; ++r) {
      for (int m = 1; m < 16; m <<= 1) sm[r] += __shfl_xor(sm[r], m, 32);
      sm[r] = 1.f / sm[r];
    }
#pragma unroll
    for (int nt = 0; nt < NKT; ++nt)
#pragma unroll
      for (int r = 0; r < 8; ++r)
        Ps[w][(r + half * 8) * NKA + nt * 16 + col] = (_Float16)(s[nt][r] * sm[r]);
#pragma unroll
    for (int ktv = 0; ktv < KTV; ++ktv) {
      v16h a = load_a_frag(&Ps[w][0] + ktv * 32, NKA);
#pragma unroll
      for (int ntv = 0; ntv < 4; ++ntv) {
        v16h bf = *(const v16h*)(Vp + ((long)b * (2 * KTV * 4) + (long)(h * KTV + ktv) * 4 + ntv) * 512 + l * 16);
        o[h * 4 + ntv] = wmma16(a, bf, o[h * 4 + ntv]);
      }
    }
  }
  // stage O into Qs (same-wave LDS, in-order)
#pragma unroll
  for (int nt = 0; nt < 8; ++nt)
#pragma unroll
    for (int r = 0; r < 8; ++r)
      Qs[w][(r + half * 8) * CH + nt * 16 + col] = (_Float16)o[nt][r];
  // output projection
  v8f y[8];
#pragma unroll
  for (int i = 0; i < 8; ++i) y[i] = vzero8();
#pragma unroll
  for (int kt = 0; kt < 4; ++kt) {
    v16h a = load_a_frag(&Qs[w][0] + kt * 32, CH);
#pragma unroll
    for (int nt = 0; nt < 8; ++nt) {
      v16h bf = *(const v16h*)(WoP + (long)(kt * 8 + nt) * 512 + l * 16);
      y[nt] = wmma16(a, bf, y[nt]);
    }
  }
#pragma unroll
  for (int nt = 0; nt < 8; ++nt) {
    float bb = bo[nt * 16 + col];
#pragma unroll
    for (int r = 0; r < 8; ++r) {
      int rr = row0 + r + half * 8;
      if (rr < M)
        Out[(long)b * oStride + (long)rr * CH + nt * 16 + col] = (_Float16)(y[nt][r] + bb);
    }
  }
}

// ---------------- connect rescat + masked log-softmax + top-k keep mask ----------------
__global__ void connect_kernel(const float* __restrict__ hlane, const _Float16* __restrict__ hatt,
                               const float* __restrict__ cvecConn, const float* __restrict__ scalConn,
                               const float* __restrict__ W1, const float* __restrict__ Wf,
                               const float* __restrict__ g_, const float* __restrict__ be_,
                               const float* __restrict__ c_mask,
                               float* __restrict__ logLsOut, float* __restrict__ lsMask) {
  int b = blockIdx.x, t = threadIdx.x;  // 128 threads
  __shared__ float hlS[CH];
  __shared__ float haS[CH];
  __shared__ float red[CH];
  __shared__ float hls[NL];
  for (int i = 0; i < NL; ++i) {
    const float* hl = hlane + ((long)b * NL + i) * CH;
    const _Float16* ha = hatt + ((long)b * NL + i) * CH;
    hlS[t] = hl[t];
    haS[t] = (float)ha[t];
    __syncthreads();
    float acc = cvecConn[b * CH + t];
    for (int k = 0; k < CH; ++k)
      acc += hlS[k] * W1[(128 + k) * CH + t] + haS[k] * W1[(256 + k) * CH + t];
    // LN
    red[t] = acc; __syncthreads();
    for (int s = 64; s > 0; s >>= 1) { if (t < s) red[t] += red[t + s]; __syncthreads(); }
    float mu = red[0] * (1.f / 128.f); __syncthreads();
    float d = acc - mu;
    red[t] = d * d; __syncthreads();
    for (int s = 64; s > 0; s >>= 1) { if (t < s) red[t] += red[t + s]; __syncthreads(); }
    float var = red[0] * (1.f / 128.f); __syncthreads();
    float h = d * rsqrtf(var + 1e-5f) * g_[t] + be_[t];
    h = h > 0.f ? h : 0.f;
    red[t] = hlS[t] * Wf[128 + t] + haS[t] * Wf[256 + t] + h * Wf[384 + t];
    __syncthreads();
    for (int s = 64; s > 0; s >>= 1) { if (t < s) red[t] += red[t + s]; __syncthreads(); }
    if (t == 0) hls[i] = red[0] + scalConn[b];
    __syncthreads();
  }
  if (t == 0) {
    float m[NL], ls[NL];
    int idx[NL];
    float mx = -3.0e38f;
    for (int i = 0; i < NL; ++i) {
      m[i] = hls[i] - 1.0e7f * (1.f - c_mask[(long)b * NM + i]);
      mx = fmaxf(mx, m[i]);
    }
    float sum = 0.f;
    for (int i = 0; i < NL; ++i) sum += __expf(m[i] - mx);
    float lse = mx + __logf(sum);
    for (int i = 0; i < NL; ++i) {
      float lg = m[i] - lse;
      logLsOut[(long)b * NL + i] = lg;
      ls[i] = __expf(lg);
      idx[i] = i;
    }
    // stable descending insertion sort (matches top_k tie ordering)
    for (int i = 1; i < NL; ++i) {
      float kv = ls[i]; int ki = idx[i];
      int j = i - 1;
      while (j >= 0 && ls[j] < kv) { ls[j + 1] = ls[j]; idx[j + 1] = idx[j]; --j; }
      ls[j + 1] = kv; idx[j + 1] = ki;
    }
    float cum = 0.f;
    for (int j = 0; j < NL; ++j) {
      cum += ls[j];
      lsMask[(long)b * NL + idx[j]] = (cum - ls[j] <= 0.95f) ? 1.f : 0.f;
    }
  }
}

// ---------------- convert rescat: heat logits ----------------
__global__ void __launch_bounds__(128) convert_kernel(
    const _Float16* __restrict__ pos1, const _Float16* __restrict__ pos2,
    const _Float16* __restrict__ pos3,
    const _Float16* __restrict__ W1P,  // packed (512 x 128)
    const float* __restrict__ cvecConv, const float* __restrict__ scalConv,
    const float* __restrict__ Wf, const float* __restrict__ g_, const float* __restrict__ be_,
    float* __restrict__ heat) {
  int b = blockIdx.y;
  int w = threadIdx.x >> 5, l = threadIdx.x & 31;
  int col = l & 15, half = l >> 4;
  int row0 = blockIdx.x * 64 + w * 16;
  __shared__ float WfS[640];
  __shared__ float gS[CH];
  __shared__ float beS[CH];
  __shared__ float Xs[4][16 * CH];
  for (int i = threadIdx.x; i < 640; i += blockDim.x) WfS[i] = Wf[i];
  for (int i = threadIdx.x; i < CH; i += blockDim.x) { gS[i] = g_[i]; beS[i] = be_[i]; }
  __syncthreads();
  if (row0 >= NC) return;

  v8f acc[8];
#pragma unroll
  for (int nt = 0; nt < 8; ++nt) {
    float c0 = cvecConv[b * CH + nt * 16 + col];
#pragma unroll
    for (int r = 0; r < 8; ++r) acc[nt][r] = c0;
  }
  float rowdot = 0.f;
  const _Float16* srcs[3] = {pos1, pos2, pos3};
  for (int s3 = 0; s3 < 3; ++s3) {
    const _Float16* base = srcs[s3] + (long)b * NC * CH;
#pragma unroll
    for (int kt = 0; kt < 4; ++kt) {
      v16h a = load_a_clamped(base, CH, row0, NC, kt * 32);
      int kb = 128 + s3 * 128 + kt * 32 + ((l >> 4) << 3);
#pragma unroll
      for (int e = 0; e < 8; ++e)
        rowdot += (float)a[e] * WfS[kb + e] + (float)a[e + 8] * WfS[kb + 16 + e];
      int ktg = 4 + s3 * 4 + kt;  // W1 rows 128.. ; rows 0..127 handled by cvecConv
#pragma unroll
      for (int nt = 0; nt < 8; ++nt) {
        v16h bf = *(const v16h*)(W1P + (long)(ktg * 8 + nt) * 512 + l * 16);
        acc[nt] = wmma16(a, bf, acc[nt]);
      }
    }
  }
  rowdot += __shfl_xor(rowdot, 16, 32);
#pragma unroll
  for (int nt = 0; nt < 8; ++nt)
#pragma unroll
    for (int r = 0; r < 8; ++r)
      Xs[w][(r + half * 8) * CH + nt * 16 + col] = acc[nt][r];
  // serial LN + final dot: lanes 0..15, one row each (same-wave LDS, in-order)
  if (l < 16) {
    int rr = row0 + l;
    if (rr < NC) {
      const float* xr = &Xs[w][l * CH];
      float mu = 0.f;
      for (int c = 0; c < CH; ++c) mu += xr[c];
      mu *= (1.f / 128.f);
      float var = 0.f;
      for (int c = 0; c < CH; ++c) { float d = xr[c] - mu; var += d * d; }
      float rstd = rsqrtf(var * (1.f / 128.f) + 1e-5f);
      float hdot = 0.f;
      for (int c = 0; c < CH; ++c) {
        float h = (xr[c] - mu) * rstd * gS[c] + beS[c];
        h = h > 0.f ? h : 0.f;
        hdot += h * WfS[512 + c];
      }
      heat[(long)b * NC + rr] = scalConv[b] + rowdot + hdot;
    }
  }
}

// ---------------- final log-softmax over 12000 ----------------
__global__ void logsoftmax_kernel(const float* __restrict__ heat, float* __restrict__ out) {
  int b = blockIdx.x, t = threadIdx.x;  // 256 threads
  __shared__ float red[256];
  float mx = -3.0e38f;
  for (int i = t; i < NC; i += 256) mx = fmaxf(mx, heat[(long)b * NC + i]);
  red[t] = mx; __syncthreads();
  for (int s = 128; s > 0; s >>= 1) { if (t < s) red[t] = fmaxf(red[t], red[t + s]); __syncthreads(); }
  mx = red[0]; __syncthreads();
  float sum = 0.f;
  for (int i = t; i < NC; i += 256) sum += __expf(heat[(long)b * NC + i] - mx);
  red[t] = sum; __syncthreads();
  for (int s = 128; s > 0; s >>= 1) { if (t < s) red[t] += red[t + s]; __syncthreads(); }
  float lse = mx + __logf(red[0]);
  __syncthreads();
  for (int i = t; i < NC; i += 256) out[(long)b * NC + i] = heat[(long)b * NC + i] - lse;
}

// ---------------- host launcher ----------------
extern "C" void kernel_launch(void* const* d_in, const int* in_sizes, int n_in,
                              void* d_out, int out_size, void* d_ws, size_t ws_size,
                              hipStream_t stream) {
  (void)in_sizes; (void)n_in; (void)out_size; (void)ws_size;
  auto F = [&](int i) { return (const float*)d_in[i]; };
  char* wsp = (char*)d_ws;
  size_t off = 0;
  auto allocH = [&](size_t n) {
    _Float16* p = (_Float16*)(wsp + off);
    off += ((n * 2 + 255) / 256) * 256;
    return p;
  };
  auto allocF = [&](size_t n) {
    float* p = (float*)(wsp + off);
    off += ((n * 4 + 255) / 256) * 256;
    return p;
  };

  // packed weights
  _Float16 *pWq_ls = allocH(16384), *pWk_ls = allocH(16384), *pWv_ls = allocH(16384), *pWo_ls = allocH(16384);
  _Float16 *pWq_2c = allocH(16384), *pWk_2c = allocH(16384), *pWv_2c = allocH(16384), *pWo_2c = allocH(16384);
  _Float16 *pWq_22 = allocH(16384), *pWk_22 = allocH(16384), *pWv_22 = allocH(16384), *pWo_22 = allocH(16384);
  _Float16* pW1cv = allocH(512 * 128);
  // per-batch prep
  float* ego = allocF(CB * CH);
  float* cvecE2c = allocF(CB * CH);
  float* cvecConv = allocF(CB * CH);
  float* cvecConn = allocF(CB * CH);
  float* scalConv = allocF(CB);
  float* scalConn = allocF(CB);
  // f16 activations
  _Float16* hlane16 = allocH((size_t)CB * NL * CH);
  _Float16* hmid16 = allocH((size_t)CB * NM * CH);
  _Float16* hatt16 = allocH((size_t)CB * NL * CH);
  // packed K/V
  _Float16* KpLS = allocH((size_t)CB * 2 * 2 * 16 * 512);
  _Float16* VpLS = allocH((size_t)CB * 2 * 8 * 4 * 512);
  _Float16* Kp2c = allocH((size_t)CB * 2 * 2 * 16 * 512);
  _Float16* Vp2c = allocH((size_t)CB * 2 * 8 * 4 * 512);
  _Float16* Kp22 = allocH((size_t)CB * 2 * 2 * 4 * 512);
  _Float16* Vp22 = allocH((size_t)CB * 2 * 2 * 4 * 512);
  // big intermediates
  _Float16* pos1 = allocH((size_t)CB * NC * CH);
  _Float16* pos2 = allocH((size_t)CB * NC * CH);
  _Float16* pos3 = allocH((size_t)CB * NC * CH);
  float* lsMask = allocF((size_t)CB * NL);
  float* heat = allocF((size_t)CB * NC);

  // 1) pack weights (K/32 * N/16 tiles, 32 lanes each)
  pack_b_kernel<<<32, 32, 0, stream>>>(F(6), pWq_ls, 128, 128);
  pack_b_kernel<<<32, 32, 0, stream>>>(F(8), pWk_ls, 128, 128);
  pack_b_kernel<<<32, 32, 0, stream>>>(F(10), pWv_ls, 128, 128);
  pack_b_kernel<<<32, 32, 0, stream>>>(F(12), pWo_ls, 128, 128);
  pack_b_kernel<<<32, 32, 0, stream>>>(F(24), pWq_2c, 128, 128);
  pack_b_kernel<<<32, 32, 0, stream>>>(F(26), pWk_2c, 128, 128);
  pack_b_kernel<<<32, 32, 0, stream>>>(F(28), pWv_2c, 128, 128);
  pack_b_kernel<<<32, 32, 0, stream>>>(F(30), pWo_2c, 128, 128);
  pack_b_kernel<<<32, 32, 0, stream>>>(F(32), pWq_22, 128, 128);
  pack_b_kernel<<<32, 32, 0, stream>>>(F(34), pWk_22, 128, 128);
  pack_b_kernel<<<32, 32, 0, stream>>>(F(36), pWv_22, 128, 128);
  pack_b_kernel<<<32, 32, 0, stream>>>(F(38), pWo_22, 128, 128);
  pack_b_kernel<<<128, 32, 0, stream>>>(F(40), pW1cv, 512, 128);

  // 2) per-batch prep
  prep_batch_kernel<<<CB, 128, 0, stream>>>(F(2), F(20), F(21), F(40), F(41), F(44), F(45),
                                            F(14), F(15), F(18), F(19),
                                            ego, cvecE2c, cvecConv, cvecConn, scalConv, scalConn);
  // 3) f16 conversions
  f32_to_f16_kernel<<<(CB * NL * CH + 255) / 256, 256, 0, stream>>>(F(0), hlane16, (long)CB * NL * CH);
  f32_to_f16_kernel<<<(CB * NM * CH + 255) / 256, 256, 0, stream>>>(F(1), hmid16, (long)CB * NM * CH);

  // 4) ls_attn K/V + attention -> h_att
  kvproj_kernel<16><<<dim3(4, CB), 128, 0, stream>>>(hmid16, (long)NM * CH, NM,
                                                     pWk_ls, F(9), pWv_ls, F(11), KpLS, VpLS);
  attn_kernel<16><<<dim3(1, CB), 128, 0, stream>>>(hlane16, (long)NL * CH, NL,
                                                   pWq_ls, F(7), KpLS, VpLS,
                                                   F(4), NM, NM, pWo_ls, F(13),
                                                   hatt16, (long)NL * CH);
  // 5) connect -> log_ls (d_out[0:880]) + ls_mask
  connect_kernel<<<CB, 128, 0, stream>>>(F(0), hatt16, cvecConn, scalConn,
                                         F(14), F(18), F(16), F(17), F(4),
                                         (float*)d_out, lsMask);
  // 6) pos1
  pos1_kernel<<<(CB * NC) / 8, 256, 0, stream>>>(F(3), cvecE2c, F(20), F(22), F(23), pos1);

  // 7) l2c / l2c2 K/V
  kvproj_kernel<16><<<dim3(4, CB), 128, 0, stream>>>(hmid16, (long)NM * CH, NM,
                                                     pWk_2c, F(27), pWv_2c, F(29), Kp2c, Vp2c);
  kvproj_kernel<4><<<dim3(1, CB), 128, 0, stream>>>(hlane16, (long)NL * CH, NL,
                                                    pWk_22, F(35), pWv_22, F(37), Kp22, Vp22);
  // 8) big attentions
  attn_kernel<16><<<dim3(188, CB), 128, 0, stream>>>(pos1, (long)NC * CH, NC,
                                                     pWq_2c, F(25), Kp2c, Vp2c,
                                                     F(4), NM, NM, pWo_2c, F(31),
                                                     pos2, (long)NC * CH);
  attn_kernel<4><<<dim3(188, CB), 128, 0, stream>>>(pos1, (long)NC * CH, NC,
                                                    pWq_22, F(33), Kp22, Vp22,
                                                    lsMask, NL, NL, pWo_22, F(39),
                                                    pos3, (long)NC * CH);
  // 9) convert -> heat logits
  convert_kernel<<<dim3(188, CB), 128, 0, stream>>>(pos1, pos2, pos3, pW1cv,
                                                    cvecConv, scalConv, F(44), F(42), F(43), heat);
  // 10) log-softmax -> d_out[880:]
  logsoftmax_kernel<<<CB, 256, 0, stream>>>(heat, (float*)d_out + CB * NL);
}
